// SMConvBlock_4286377361803
// MI455X (gfx1250) — matrix operand order repaired and checked by reading, not compiled, not run on hardware
//
#include <hip/hip_runtime.h>
#include <hip/hip_bf16.h>

typedef __attribute__((ext_vector_type(16))) _Float16 v16h;
typedef __attribute__((ext_vector_type(8)))  float    v8f;

// Tile geometry
#define TS   16      // output tile 16x16 pixels
#define HT   18      // h buffer (tile + halo 1)
#define IT   20      // image buffer (tile + halo 2)
#define C1   20      // hidden channels of kernel-net
#define CP   32      // channel dim padded to one WMMA K-step

struct Smem {
    alignas(64) _Float16 h[HT][HT][CP];     // conv1 activations, channel-last (B operand)
    alignas(64) _Float16 aFrag[9][32][16];  // conv2 weights pre-packed per tap in A-frag layout
    float    img[IT][IT];                   // image tile, halo 2
    float    xt[4][HT][HT];                 // x tile per in-channel, halo 1
    float    w1[C1][9];                     // conv1 weights (current p)
    float    b1[C1];
    float    b2[16];                        // conv2 bias padded
    float    yacc[4][TS][TS];               // output accumulator (4 out channels)
};

__global__ __launch_bounds__(256)
void smconv_fused_kernel(const float* __restrict__ image,
                         const float* __restrict__ x,
                         const float* __restrict__ W1,
                         const float* __restrict__ b1,
                         const float* __restrict__ W2,
                         const float* __restrict__ b2,
                         float* __restrict__ out)
{
    __shared__ Smem s;

    const int tid  = threadIdx.x;
    const int lane = tid & 31;
    const int wave = tid >> 5;            // 0..7
    const int grp  = lane >> 4;           // half-wave group
    const int lm   = lane & 15;           // A-row (ko) / B-col (pixel-in-ntile)

    const int blk  = blockIdx.x;          // 0..1023
    const int b    = blk >> 8;            // batch
    const int trem = blk & 255;
    const int gy0  = (trem >> 4) * TS;
    const int gx0  = (trem & 15) * TS;

    // ---- stage image tile (halo 2, zero pad = 'SAME' through two stacked 3x3) ----
    for (int idx = tid; idx < IT * IT; idx += 256) {
        int iy = idx / IT, ix = idx % IT;
        int gy = gy0 + iy - 2, gx = gx0 + ix - 2;
        float v = 0.f;
        if ((unsigned)gy < 256u && (unsigned)gx < 256u)
            v = image[(b * 256 + gy) * 256 + gx];
        s.img[iy][ix] = v;
    }
    // ---- stage x tile (4 channels, halo 1) ----
    for (int idx = tid; idx < 4 * HT * HT; idx += 256) {
        int j = idx / (HT * HT), r = idx % (HT * HT);
        int iy = r / HT, ix = r % HT;
        int gy = gy0 + iy - 1, gx = gx0 + ix - 1;
        float v = 0.f;
        if ((unsigned)gy < 256u && (unsigned)gx < 256u)
            v = x[((b * 4 + j) * 256 + gy) * 256 + gx];
        s.xt[j][iy][ix] = v;
    }
    // ---- zero y accumulator and the padded channels of h (once; never rewritten) ----
    for (int idx = tid; idx < 4 * TS * TS; idx += 256)
        ((float*)s.yacc)[idx] = 0.f;
    for (int idx = tid; idx < (int)(sizeof(s.h) / 4); idx += 256)
        ((unsigned*)s.h)[idx] = 0u;
    __syncthreads();

    for (int p = 0; p < 16; ++p) {
        const int i_out = p >> 2;   // nc_out index
        const int j_in  = p & 3;    // nc_in index

        // ---- conv1 weights into LDS ----
        for (int idx = tid; idx < C1 * 9; idx += 256)
            s.w1[idx / 9][idx % 9] = W1[p * C1 * 9 + idx];
        if (tid < C1) s.b1[tid] = b1[p * C1 + tid];
        if (tid < 16) s.b2[tid] = (tid < 9) ? b2[p * 9 + tid] : 0.f;

        // ---- pre-pack conv2 weights per tap into A-fragment layout (ISA 7.12.2) ----
        // A for tap t: A[ko][ci]; lane (grp,m): hh -> ci = 16*grp + hh (contiguous k, same as B).
        for (int idx = tid; idx < 9 * 32 * 16; idx += 256) {
            int t  = idx >> 9;            // tap 0..8
            int l  = (idx >> 4) & 31;     // lane
            int hh = idx & 15;            // half index within fragment
            int g  = l >> 4, m = l & 15;
            int ci = 16 * g + hh;
            float w = 0.f;
            if (m < 9 && ci < C1)
                w = W2[(p * 9 + m) * (C1 * 9) + ci * 9 + t];
            s.aFrag[t][l][hh] = (_Float16)w;
        }
        __syncthreads();

        // ---- conv1 on VALU fp32, store channel-last f16 (channels 0..19 only) ----
        for (int idx = tid; idx < C1 * HT * HT; idx += 256) {
            int c = idx / (HT * HT), r = idx % (HT * HT);
            int hy = r / HT, hx = r % HT;
            float acc = s.b1[c];
#pragma unroll
            for (int dy = 0; dy < 3; ++dy)
#pragma unroll
                for (int dx = 0; dx < 3; ++dx)
                    acc += s.img[hy + dy][hx + dx] * s.w1[c][dy * 3 + dx];
            s.h[hy][hx][c] = (_Float16)acc;
        }
        __syncthreads();

        // ---- conv2: 9 tap-GEMMs, K=32 (20 real ch), all-contiguous fragment loads ----
        // Wave handles n-tiles 2*wave, 2*wave+1 => pixel rows ly0, ly1; col = lm.
        const int ly0 = wave * 2, ly1 = wave * 2 + 1;
        v8f c0 = {}, c1 = {};
#pragma unroll
        for (int t = 0; t < 9; ++t) {
            const int dy = t / 3, dx = t % 3;
            v16h a  = *(const v16h*)&s.aFrag[t][lane][0];
            v16h bA = *(const v16h*)&s.h[ly0 + dy][lm + dx][grp * 16];
            v16h bB = *(const v16h*)&s.h[ly1 + dy][lm + dx][grp * 16];
            c0 = __builtin_amdgcn_wmma_f32_16x16x32_f16(false, a, false, bA,
                                                        (short)0, c0, false, false);
            c1 = __builtin_amdgcn_wmma_f32_16x16x32_f16(false, a, false, bB,
                                                        (short)0, c1, false, false);
        }

        // ---- fuse bias + einsum with x patches, accumulate into y tile ----
#pragma unroll
        for (int half = 0; half < 2; ++half) {
            const int ly = half ? ly1 : ly0;
            const v8f& cf = half ? c1 : c0;
            float part = 0.f;
#pragma unroll
            for (int r = 0; r < 8; ++r) {
                int M = r + 8 * grp;          // conv2 out channel = patch tap
                if (M < 9) {
                    float kv = cf[r] + s.b2[M];
                    part += kv * s.xt[j_in][ly + M / 3][lm + M % 3];
                }
            }
            atomicAdd(&s.yacc[i_out][ly][lm], part);
        }
        __syncthreads();
    }

    // ---- write y tile: out(BS, 4, 256, 256) ----
    for (int idx = tid; idx < 4 * TS * TS; idx += 256) {
        int i = idx >> 8, r = idx & 255;
        int ly = r >> 4, lx = r & 15;
        out[((b * 4 + i) * 256 + (gy0 + ly)) * 256 + (gx0 + lx)] = s.yacc[i][ly][lx];
    }
}

extern "C" void kernel_launch(void* const* d_in, const int* in_sizes, int n_in,
                              void* d_out, int out_size, void* d_ws, size_t ws_size,
                              hipStream_t stream) {
    (void)in_sizes; (void)n_in; (void)out_size; (void)d_ws; (void)ws_size;
    const float* image = (const float*)d_in[0];
    const float* x     = (const float*)d_in[1];
    const float* W1    = (const float*)d_in[2];
    const float* b1    = (const float*)d_in[3];
    const float* W2    = (const float*)d_in[4];
    const float* b2    = (const float*)d_in[5];
    float* out = (float*)d_out;

    // 4 batches x 16x16 tiles of 16x16 pixels = 1024 blocks, 8 wave32 each
    smconv_fused_kernel<<<dim3(4 * 16 * 16), dim3(256), 0, stream>>>(
        image, x, W1, b1, W2, b2, out);
}
